// VQ_61400852463739
// MI455X (gfx1250) — matrix operand order repaired and checked by reading, not compiled, not run on hardware
//
#include <hip/hip_runtime.h>
#include <hip/hip_bf16.h>

// ---------------------------------------------------------------------------
// VQ-VAE EMA update, MI455X (gfx1250, wave32).
// Dominant stage: fused (-2*xn)@E^T + ||e||^2 GEMM with in-register argmin,
// run on v_wmma_f32_16x16x32_f16 (f32 accumulate). Avoids the 2.15 GB
// distance matrix entirely (roofline: that alone would cost ~180us at
// 23.3 TB/s, vs ~70 GFLOP of f16 WMMA work).
// Round-1 changes: scalarized argmin state (kills the s_or_b32/cndmask web
// the vector selects generated) and double-buffered B tiles so loads for
// tile n+1 overlap the 8 WMMAs + epilogue of tile n.
// ---------------------------------------------------------------------------

typedef __attribute__((ext_vector_type(16))) _Float16 v16h;
typedef __attribute__((ext_vector_type(8)))  float    v8f;

#define DDIM   64
#define KEMB   8192
#define NROWS  65536
#define DECAY  0.99f
#define EPSV   1e-5f

// ---------------- ws layout (float indices into d_ws) ----------------------
#define WS_SUM     0
#define WS_SUMSQ   64
#define WS_MEAN    128
#define WS_INVSTD  192
#define WS_RSTD    256
#define WS_COUNTS  320
#define WS_E2      (WS_COUNTS + KEMB)                 // 8512
#define WS_DW      (WS_E2 + KEMB)                     // 16704
#define WS_IDX     (WS_DW + KEMB * DDIM)              // 540992
#define WS_A16     (WS_IDX + NROWS)                   // 606528
#define WS_E16     (WS_A16 + (NROWS * DDIM) / 2)      // 2703680
#define WS_ZERO_N  WS_IDX                             // zero sum..dw

// ---------------------------------------------------------------------------
__global__ void vq_zero(float* __restrict__ p, int n) {
  for (int i = blockIdx.x * blockDim.x + threadIdx.x; i < n;
       i += gridDim.x * blockDim.x)
    p[i] = 0.0f;
}

// Partial column sums / sumsq (x is [N][64] row-major; thread owns one d).
__global__ __launch_bounds__(256) void vq_stats_partial(
    const float* __restrict__ x, float* __restrict__ sum,
    float* __restrict__ sumsq, int N) {
  __shared__ float ls[256], ls2[256];
  const int d    = threadIdx.x & 63;
  const int rsub = threadIdx.x >> 6;          // 0..3
  float s = 0.0f, s2 = 0.0f;
  for (int row = blockIdx.x * 4 + rsub; row < N; row += gridDim.x * 4) {
    float v = x[(size_t)row * DDIM + d];
    s += v; s2 += v * v;
  }
  ls[threadIdx.x] = s; ls2[threadIdx.x] = s2;
  __syncthreads();
  if (rsub == 0) {
    s  = ls[d]  + ls[64 + d]  + ls[128 + d]  + ls[192 + d];
    s2 = ls2[d] + ls2[64 + d] + ls2[128 + d] + ls2[192 + d];
    atomicAdd(&sum[d], s);
    atomicAdd(&sumsq[d], s2);
  }
}

__global__ void vq_stats_finalize(float* __restrict__ w, int N) {
  int d = threadIdx.x;                        // 64 threads
  float s = w[WS_SUM + d], s2 = w[WS_SUMSQ + d];
  float mean = s / (float)N;
  float var  = s2 / (float)N - mean * mean;   // biased
  w[WS_MEAN   + d] = mean;
  w[WS_INVSTD + d] = rsqrtf(var + EPSV);
  float uv = var * ((float)N / (float)(N - 1));
  w[WS_RSTD + d] = sqrtf(uv + EPSV);
}

// A = f16(-2 * xn)  (pre-scaled so WMMA score = ||e||^2 - 2 xn.e directly)
__global__ void vq_prep_a(const float* __restrict__ x,
                          const float* __restrict__ mean,
                          const float* __restrict__ invstd,
                          _Float16* __restrict__ a16, int total) {
  for (int i = blockIdx.x * blockDim.x + threadIdx.x; i < total;
       i += gridDim.x * blockDim.x) {
    int d = i & 63;
    float xn = (x[i] - mean[d]) * invstd[d];
    a16[i] = (_Float16)(-2.0f * xn);
  }
}

// E16 = f16(E); e2[k] = ||E_k||^2 (f32)
__global__ void vq_prep_b(const float* __restrict__ emb,
                          _Float16* __restrict__ e16,
                          float* __restrict__ e2, int K) {
  const int tid = blockIdx.x * blockDim.x + threadIdx.x;
  const int total = K * DDIM;
  for (int i = tid; i < total; i += gridDim.x * blockDim.x)
    e16[i] = (_Float16)emb[i];
  for (int k = tid; k < K; k += gridDim.x * blockDim.x) {
    float s = 0.0f;
    #pragma unroll 8
    for (int d = 0; d < DDIM; ++d) { float v = emb[k * DDIM + d]; s += v * v; }
    e2[k] = s;
  }
}

// ---------------------------------------------------------------------------
// Fused distance GEMM + argmin.  One wave owns 64 rows (4 WMMA row-blocks),
// streams all 512 code tiles with ping-pong B buffers; 16 wmma per loop body.
// A/B fragment loads follow the ISA 16-bit operand layouts: two contiguous
// b128 loads per lane.
// ---------------------------------------------------------------------------
__global__ __launch_bounds__(256) void vq_argmin_wmma(
    const _Float16* __restrict__ A,   // [N][64]  (= -2*xn, f16)
    const _Float16* __restrict__ B,   // [K][64]  (codebook, f16)
    const float*    __restrict__ e2,  // [K]
    int* __restrict__ outIdx) {
  const int lane   = threadIdx.x & 31;
  const int wave   = (blockIdx.x * blockDim.x + threadIdx.x) >> 5;
  const int rowB   = wave * 64;
  const int laneHi = lane >> 4;       // half-wave select
  const int lane15 = lane & 15;

  // A fragments: row-block r, K-chunk kc (kc*32).
  // lanes 0-15 : row=lane15, elems 0-7 = K[kc*32+0..7], 8-15 = K[kc*32+16..23]
  // lanes 16-31: row=lane15, elems 0-7 = K[kc*32+8..15], 8-15 = K[kc*32+24..31]
  v16h a[4][2];
  #pragma unroll
  for (int r = 0; r < 4; ++r) {
    const size_t row = (size_t)(rowB + r * 16 + lane15);
    #pragma unroll
    for (int kc = 0; kc < 2; ++kc) {
      const _Float16* p = A + row * DDIM + kc * 32 + laneHi * 8;
      ((uint4*)&a[r][kc])[0] = *(const uint4*)(p);
      ((uint4*)&a[r][kc])[1] = *(const uint4*)(p + 16);
    }
  }

  // Scalar argmin state (fully unrolled -> registers; avoids vector
  // insert-element select chains).
  float bestv[4][8];
  int   besti[4][8];
  #pragma unroll
  for (int r = 0; r < 4; ++r)
    #pragma unroll
    for (int e = 0; e < 8; ++e) { bestv[r][e] = 3.4e38f; besti[r][e] = 0; }

  // Ping-pong B-tile buffers (32x16 = E^T tile): lanes 0-15 hold K=0..7,16..23 of
  // column `col` (row-major E), lanes 16-31 hold K=8..15,24..31.
  v16h  bf[2][2];
  float e2v[2];

  auto loadB = [&](int cb, int buf) {
    const int col = cb + lane15;
    #pragma unroll
    for (int kc = 0; kc < 2; ++kc) {
      const _Float16* p = B + (size_t)col * DDIM + kc * 32 + laneHi * 8;
      ((uint4*)&bf[buf][kc])[0] = *(const uint4*)(p);
      ((uint4*)&bf[buf][kc])[1] = *(const uint4*)(p + 16);
    }
    e2v[buf] = e2[col];
  };

  auto compute = [&](int cb, int buf) {
    const int col = cb + lane15;
    v8f c0;
    #pragma unroll
    for (int e = 0; e < 8; ++e) c0[e] = e2v[buf];  // column-constant ||e||^2
    #pragma unroll
    for (int r = 0; r < 4; ++r) {
      v8f acc = __builtin_amdgcn_wmma_f32_16x16x32_f16(
          false, a[r][0], false, bf[buf][0], (short)0, c0, false, false);
      acc = __builtin_amdgcn_wmma_f32_16x16x32_f16(
          false, a[r][1], false, bf[buf][1], (short)0, acc, false, false);
      #pragma unroll
      for (int e = 0; e < 8; ++e) {
        const float av = acc[e];
        const bool  lt = av < bestv[r][e];
        bestv[r][e] = lt ? av  : bestv[r][e];
        besti[r][e] = lt ? col : besti[r][e];
      }
    }
  };

  loadB(0, 0);
  for (int cb = 0; cb < KEMB; cb += 32) {
    loadB(cb + 16, 1);
    compute(cb, 0);
    if (cb + 32 < KEMB) loadB(cb + 32, 0);
    compute(cb + 16, 1);
  }

  // Per-row min across the 16 lanes of each half (C/D layout: lanes 0-15 hold
  // rows 0-7 in elems 0-7; lanes 16-31 hold rows 8-15). xor masks 1,2,4,8
  // stay within each half-wave.
  #pragma unroll
  for (int r = 0; r < 4; ++r) {
    #pragma unroll
    for (int e = 0; e < 8; ++e) {
      float v = bestv[r][e]; int id = besti[r][e];
      #pragma unroll
      for (int m = 1; m < 16; m <<= 1) {
        float ov = __shfl_xor(v, m, 32);
        int   oi = __shfl_xor(id, m, 32);
        if (ov < v || (ov == v && oi < id)) { v = ov; id = oi; }
      }
      besti[r][e] = id;
    }
    if (lane15 == 0) {
      #pragma unroll
      for (int e = 0; e < 8; ++e)
        outIdx[rowB + r * 16 + laneHi * 8 + e] = besti[r][e];
    }
  }
}

// One wave per row: segment-sum scatter with f32 global atomics.
__global__ __launch_bounds__(256) void vq_scatter(
    const float* __restrict__ x, const float* __restrict__ mean,
    const float* __restrict__ invstd, const int* __restrict__ idx,
    float* __restrict__ dw, float* __restrict__ counts, int N) {
  const int wave = (blockIdx.x * blockDim.x + threadIdx.x) >> 5;
  const int lane = threadIdx.x & 31;
  if (wave >= N) return;
  const int k = idx[wave];
  const size_t xb = (size_t)wave * DDIM;
  float xn0 = (x[xb + lane]      - mean[lane])      * invstd[lane];
  float xn1 = (x[xb + lane + 32] - mean[lane + 32]) * invstd[lane + 32];
  atomicAdd(&dw[(size_t)k * DDIM + lane],      xn0);
  atomicAdd(&dw[(size_t)k * DDIM + lane + 32], xn1);
  if (lane == 0) atomicAdd(&counts[k], 1.0f);
}

// EMA update + BN un-normalization; writes all three outputs.
__global__ void vq_finalize(const float* __restrict__ emb,
                            const float* __restrict__ cs,
                            const float* __restrict__ dw,
                            const float* __restrict__ counts,
                            const float* __restrict__ mean,
                            const float* __restrict__ rstd,
                            float* __restrict__ out, int K) {
  const int total = K * DDIM;
  for (int i = blockIdx.x * blockDim.x + threadIdx.x; i < total;
       i += gridDim.x * blockDim.x) {
    int k = i >> 6, d = i & 63;
    float ns = cs[k] * DECAY + (1.0f - DECAY) * counts[k];
    float ne = (cs[k] * emb[i] * DECAY + (1.0f - DECAY) * dw[i]) / ns;
    out[i]         = ne * rstd[d] + mean[d];   // vq_embedding_output
    out[total + i] = ne;                       // new_emb
    if (d == 0) out[2 * total + k] = ns;       // new_size
  }
}

// ---------------------------------------------------------------------------
extern "C" void kernel_launch(void* const* d_in, const int* in_sizes, int n_in,
                              void* d_out, int out_size, void* d_ws,
                              size_t ws_size, hipStream_t stream) {
  const float* x   = (const float*)d_in[0];   // [N,64]
  const float* emb = (const float*)d_in[1];   // [K,64]
  const float* cs  = (const float*)d_in[2];   // [K]
  float* out = (float*)d_out;

  const int N = in_sizes[0] / DDIM;           // 65536
  const int K = in_sizes[2];                  // 8192

  float*     w      = (float*)d_ws;
  float*     sum    = w + WS_SUM;
  float*     sumsq  = w + WS_SUMSQ;
  float*     mean   = w + WS_MEAN;
  float*     invstd = w + WS_INVSTD;
  float*     rstd   = w + WS_RSTD;
  float*     counts = w + WS_COUNTS;
  float*     e2     = w + WS_E2;
  float*     dw     = w + WS_DW;
  int*       idx    = (int*)(w + WS_IDX);
  _Float16*  a16    = (_Float16*)(w + WS_A16);
  _Float16*  e16    = (_Float16*)(w + WS_E16);

  vq_zero<<<512, 256, 0, stream>>>(w, WS_ZERO_N);
  vq_stats_partial<<<256, 256, 0, stream>>>(x, sum, sumsq, N);
  vq_stats_finalize<<<1, 64, 0, stream>>>(w, N);
  vq_prep_a<<<2048, 256, 0, stream>>>(x, mean, invstd, a16, N * DDIM);
  vq_prep_b<<<256, 256, 0, stream>>>(emb, e16, e2, K);
  // 65536 rows / 64 rows-per-wave = 1024 waves = 128 blocks of 8 waves
  vq_argmin_wmma<<<(N / 64) / 8, 256, 0, stream>>>(a16, e16, e2, idx);
  vq_scatter<<<(N * 32 + 255) / 256, 256, 0, stream>>>(x, mean, invstd, idx,
                                                       dw, counts, N);
  vq_finalize<<<1024, 256, 0, stream>>>(emb, cs, dw, counts, mean, rstd, out,
                                        K);
}
